// AUG_47210280517876
// MI455X (gfx1250) — matrix operand order repaired and checked by reading, not compiled, not run on hardware
//
#include <hip/hip_runtime.h>
#include <hip/hip_bf16.h>

// ---------------------------------------------------------------------------
// CDNA5 (gfx1250): GEMMs via v_wmma_f32_16x16x32_bf16 with async-to-LDS
// double-buffered B-strip staging (global_load_async_to_lds_b128 + ASYNCcnt).
// ---------------------------------------------------------------------------

typedef __attribute__((ext_vector_type(16))) __bf16 v16bf;
typedef __attribute__((ext_vector_type(8)))  float  v8f;

#define BB 16384  // batch

struct __align__(8) Bf16x4 { __bf16 a, b, c, d; };

// ---- fp32 -> bf16 conversion, 4 elements / thread --------------------------
__global__ void f32_to_bf16_4(const float* __restrict__ in,
                              __bf16* __restrict__ out, long n4) {
  long i = blockIdx.x * (long)blockDim.x + threadIdx.x;
  if (i >= n4) return;
  float4 v = ((const float4*)in)[i];
  Bf16x4 o{(__bf16)v.x, (__bf16)v.y, (__bf16)v.z, (__bf16)v.w};
  ((Bf16x4*)out)[i] = o;
}

// ---- WMMA GEMM: Y[M,N] = X[M,K] @ W[N,K]^T + bias --------------------------
// Block = 8 waves -> 128x64 output supertile. Wave w owns M-tile (16 rows),
// 4 accumulators across a shared 64-column N strip. The B strip (64 rows x
// 64 bytes per 32-wide K-chunk = 4KB) is staged in LDS, double-buffered,
// copied with global_load_async_to_lds_b128 (each of 256 lanes moves 16B).
//
// A-frag (16x32 bf16): lane<16 -> row=lane, bytes [0,16)+[32,48);
//                      lane>=16 -> row=lane-16, bytes [16,32)+[48,64).
// B-frag (32x16 bf16): column n = W row (contiguous K); lane n reads bytes
//                      [hi*16, hi*16+16) and +32 of its 64B LDS row.
template <bool RELU, bool OBF16>
__global__ void gemm_wmma_async(const __bf16* __restrict__ X, int ldx,
                                const __bf16* __restrict__ W,  // N rows x K
                                const float* __restrict__ bias,
                                void* __restrict__ Y, int ldy, int colOff,
                                int N, int K) {
  __shared__ __align__(16) char bufB[2][64 * 64];  // 2 x 4KB K-chunks

  const int lane = threadIdx.x & 31;
  const int wave = threadIdx.x >> 5;
  const int tN = blockIdx.x << 6;            // 64-col strip
  const int tM = (blockIdx.y << 7) + (wave << 4);
  const int r  = lane & 15;
  const int hi = lane >> 4;

  // async-copy mapping: 256 threads x 16B cover the 64x64B strip chunk
  const int crow = threadIdx.x >> 2;         // 0..63 (N row of strip)
  const int cseg = threadIdx.x & 3;          // 16B segment within 64B row
  const char* gB = (const char*)(W + (size_t)(tN + crow) * K) + cseg * 16;
  const unsigned ldsB0 =
      (unsigned)(uintptr_t)(&bufB[0][0]) + crow * 64 + cseg * 16;
  const unsigned ldsB1 =
      (unsigned)(uintptr_t)(&bufB[1][0]) + crow * 64 + cseg * 16;

  v8f acc[4];
#pragma unroll
  for (int j = 0; j < 4; ++j) {
    const float bv = bias[tN + (j << 4) + r];   // bias indexed by N column
#pragma unroll
    for (int i = 0; i < 8; ++i) acc[j][i] = bv;
  }

  const char* pA = (const char*)(X + (size_t)(tM + r) * ldx) + hi * 16;
  const int steps = K >> 5;

  // prologue: stage K-chunk 0 into buffer 0
  asm volatile("global_load_async_to_lds_b128 %0, %1, off"
               :: "v"(ldsB0), "v"(gB) : "memory");
  gB += 64;

  for (int i = 0; i < steps; ++i) {
    asm volatile("s_wait_asynccnt 0x0" ::: "memory");  // own slice landed
    __syncthreads();                                   // whole chunk published
    if (i + 1 < steps) {                               // issue-ahead next chunk
      const unsigned dst = ((i + 1) & 1) ? ldsB1 : ldsB0;
      asm volatile("global_load_async_to_lds_b128 %0, %1, off"
                   :: "v"(dst), "v"(gB) : "memory");
      gB += 64;
    }
    union { uint4 u[2]; v16bf v; } a;
    a.u[0] = *(const uint4*)(pA);
    a.u[1] = *(const uint4*)(pA + 32);
    __builtin_prefetch(pA + 512, 0, 1);                // global_prefetch_b8
    pA += 64;

    const char* sb = bufB[i & 1];
#pragma unroll
    for (int j = 0; j < 4; ++j) {
      union { uint4 u[2]; v16bf v; } b;
      const char* p = sb + ((j << 4) + r) * 64 + hi * 16;
      b.u[0] = *(const uint4*)(p);                      // ds_load_b128
      b.u[1] = *(const uint4*)(p + 32);
      acc[j] = __builtin_amdgcn_wmma_f32_16x16x32_bf16(
          false, a.v, false, b.v, (short)0, acc[j], false, false);
    }
    __syncthreads();   // all waves done reading buf[cur] before its rewrite
  }

#pragma unroll
  for (int j = 0; j < 4; ++j) {
#pragma unroll
    for (int i = 0; i < 8; ++i) {
      float v = acc[j][i];
      if (RELU) v = fmaxf(v, 0.f);
      const size_t row = (size_t)(tM + hi * 8 + i);
      const size_t col = (size_t)(tN + (j << 4) + r) + colOff;
      if (OBF16) ((__bf16*)Y)[row * ldy + col] = (__bf16)v;
      else       ((float*)Y)[row * ldy + col] = v;
    }
  }
}

// ---- out = softmax(Q*K*scale over 32-wide head) * V  (elementwise attn) ----
template <typename OT>
__global__ void softmax_mul32(const __bf16* __restrict__ Q,
                              const __bf16* __restrict__ Km,
                              const __bf16* __restrict__ V,
                              OT* __restrict__ out, int ldio, int ldo, int colOff,
                              float scale, int H, long nGroups) {
  long g = blockIdx.x * (long)blockDim.x + threadIdx.x;
  if (g >= nGroups) return;
  long b = g / H;
  int  h = (int)(g % H);
  long base = b * (long)ldio + h * 32;
  float x[32];
  float m = -1e30f;
#pragma unroll
  for (int j = 0; j < 32; ++j) {
    float v = (float)Q[base + j] * (float)Km[base + j] * scale;
    x[j] = v;
    m = fmaxf(m, v);
  }
  float s = 0.f;
#pragma unroll
  for (int j = 0; j < 32; ++j) { float e = __expf(x[j] - m); x[j] = e; s += e; }
  float inv = 1.f / s;
  long ob = b * (long)ldo + colOff + h * 32;
#pragma unroll
  for (int j = 0; j < 32; ++j)
    out[ob + j] = (OT)(x[j] * inv * (float)V[base + j]);
}

// ---- copy a bf16 column block between 2D buffers ---------------------------
__global__ void copy_cols(const __bf16* __restrict__ src, int lds, int scol,
                          __bf16* __restrict__ dst, int ldd, int dcol,
                          int ncol, long rows) {
  long i = blockIdx.x * (long)blockDim.x + threadIdx.x;
  long n = rows * ncol;
  if (i >= n) return;
  long r = i / ncol;
  int  c = (int)(i % ncol);
  dst[r * (long)ldd + dcol + c] = src[r * (long)lds + scol + c];
}

// ---- LayerNorm over n=256 per row ------------------------------------------
__global__ void layernorm_rows(const float* __restrict__ X,
                               const float* __restrict__ g,
                               const float* __restrict__ bta,
                               __bf16* __restrict__ out, int n, long rows) {
  long r = blockIdx.x * (long)blockDim.x + threadIdx.x;
  if (r >= rows) return;
  const float* x = X + r * (long)n;
  float mu = 0.f;
  for (int i = 0; i < n; ++i) mu += x[i];
  mu /= n;
  float var = 0.f;
  for (int i = 0; i < n; ++i) { float d = x[i] - mu; var += d * d; }
  var /= n;
  float rs = rsqrtf(var + 1e-6f);
  __bf16* o = out + r * (long)n;
  for (int i = 0; i < n; ++i)
    o[i] = (__bf16)((x[i] - mu) * rs * g[i] + bta[i]);
}

// ---- final scalar head: logits[r] = dot64(h1out[r], h2_w) + h2_b -----------
__global__ void head_logits(const float* __restrict__ H1,
                            const float* __restrict__ w,
                            const float* __restrict__ b,
                            float* __restrict__ out, long rows) {
  long r = blockIdx.x * (long)blockDim.x + threadIdx.x;
  if (r >= rows) return;
  const float* x = H1 + r * 64;
  float s = b[0];
#pragma unroll
  for (int i = 0; i < 64; ++i) s += x[i] * w[i];
  out[r] = s;
}

// ---------------------------------------------------------------------------

extern "C" void kernel_launch(void* const* d_in, const int* in_sizes, int n_in,
                              void* d_out, int out_size, void* d_ws, size_t ws_size,
                              hipStream_t stream) {
  (void)in_sizes; (void)n_in; (void)out_size; (void)ws_size;
  enum {
    I_META = 0, I_TEXT, I_IMAGE, I_BETA,
    I_AE1W, I_AE1B, I_AE2W, I_AE2B, I_CI1W, I_CI1B, I_CI2W, I_CI2B,
    I_FLW, I_FLB, I_SLW, I_SLB, I_SVW, I_SVB, I_INVW, I_INVB,
    I_QTW, I_QTB, I_KTW, I_KTB, I_VTW, I_VTB,
    I_QIW, I_QIB, I_KIW, I_KIB, I_VIW, I_VIB,
    I_QMW, I_QMB, I_KMW, I_KMB, I_VMW, I_VMB,
    I_QSW, I_QSB, I_KSW, I_KSB, I_VSW, I_VSB,
    I_LNG, I_LNB, I_H1W, I_H1B, I_H2W, I_H2B,
    I_TRW, I_TRB, I_IRW, I_IRB
  };
  char* ws = (char*)d_ws;
  size_t off = 0;
  auto alloc = [&](size_t bytes) -> void* {
    void* p = ws + off;
    off += (bytes + 255) & ~(size_t)255;
    return p;
  };
  auto cvt_to = [&](const float* src, __bf16* dst, size_t n) {
    long n4 = (long)(n / 4);
    f32_to_bf16_4<<<(unsigned)((n4 + 255) / 256), 256, 0, stream>>>(src, dst, n4);
  };
  auto cvtW = [&](int idx, size_t n) -> __bf16* {
    __bf16* d = (__bf16*)alloc(n * 2);
    cvt_to((const float*)d_in[idx], d, n);
    return d;
  };
  auto gemm = [&](const __bf16* X, int ldx, const __bf16* W, const float* bias,
                  void* Y, int ldy, int colOff, int N, int K, bool relu, bool obf) {
    dim3 grid(N / 64, BB / 128);
    if (obf) {
      if (relu) gemm_wmma_async<true, true><<<grid, 256, 0, stream>>>(X, ldx, W, bias, Y, ldy, colOff, N, K);
      else      gemm_wmma_async<false, true><<<grid, 256, 0, stream>>>(X, ldx, W, bias, Y, ldy, colOff, N, K);
    } else {
      if (relu) gemm_wmma_async<true, false><<<grid, 256, 0, stream>>>(X, ldx, W, bias, Y, ldy, colOff, N, K);
      else      gemm_wmma_async<false, false><<<grid, 256, 0, stream>>>(X, ldx, W, bias, Y, ldy, colOff, N, K);
    }
  };
  auto bias = [&](int idx) { return (const float*)d_in[idx]; };

  // ---- weights -> bf16 (row-major (N,K): row n contiguous along K) ----
  __bf16* w_ae1 = cvtW(I_AE1W, 512 * 2048);
  __bf16* w_ae2 = cvtW(I_AE2W, 2048 * 512);
  __bf16* w_ci1 = cvtW(I_CI1W, 512 * 2048);
  __bf16* w_ci2 = cvtW(I_CI2W, 256 * 512);
  __bf16* w_fl  = cvtW(I_FLW, 256 * 768);
  __bf16* w_sl  = cvtW(I_SLW, 128 * 256);
  __bf16* w_sv  = cvtW(I_SVW, 128 * 256);
  __bf16* w_inv = cvtW(I_INVW, 128 * 256);
  __bf16* w_qt  = cvtW(I_QTW, 256 * 128);
  __bf16* w_kt  = cvtW(I_KTW, 256 * 128);
  __bf16* w_vt  = cvtW(I_VTW, 256 * 128);
  __bf16* w_qi  = cvtW(I_QIW, 256 * 128);
  __bf16* w_ki  = cvtW(I_KIW, 256 * 128);
  __bf16* w_vi  = cvtW(I_VIW, 256 * 128);
  __bf16* w_qm  = cvtW(I_QMW, 128 * 128);
  __bf16* w_km  = cvtW(I_KMW, 128 * 128);
  __bf16* w_vm  = cvtW(I_VMW, 128 * 128);
  __bf16* w_qs  = cvtW(I_QSW, 128 * 768);
  __bf16* w_ks  = cvtW(I_KSW, 128 * 768);
  __bf16* w_vs  = cvtW(I_VSW, 128 * 768);
  __bf16* w_h1  = cvtW(I_H1W, 64 * 256);
  __bf16* w_tr  = cvtW(I_TRW, 256 * 256);
  __bf16* w_ir  = cvtW(I_IRW, 256 * 256);

  // ---- activation buffers (aliased to limit peak ws ~200MB) ----
  __bf16* x_img  = (__bf16*)alloc((size_t)BB * 2048 * 2);  // later t2, later qkv_s outs
  __bf16* x_txt  = (__bf16*)alloc((size_t)BB * 768 * 2);   // later sq/sk/sv
  __bf16* x_meta = (__bf16*)alloc((size_t)BB * 128 * 2);
  __bf16* t1     = (__bf16*)alloc((size_t)BB * 512 * 2);   // later t3, later mq/mk/mv
  __bf16* imgv   = (__bf16*)alloc((size_t)BB * 256 * 2);
  __bf16* txtv   = (__bf16*)alloc((size_t)BB * 256 * 2);
  __bf16* tsti   = (__bf16*)alloc((size_t)BB * 256 * 2);   // [ts | ti]
  __bf16* ispii  = (__bf16*)alloc((size_t)BB * 256 * 2);   // [isp | ii]
  __bf16* statb  = (__bf16*)alloc((size_t)BB * 768 * 2);   // [out_t|out_i|ti|ii]
  float*  outbuf = (float*)alloc((size_t)BB * 256 * 4);    // [out_m|out_s]
  __bf16* normed = (__bf16*)alloc((size_t)BB * 256 * 2);
  float*  h1out  = (float*)alloc((size_t)BB * 64 * 4);

  cvt_to((const float*)d_in[I_IMAGE], x_img, (size_t)BB * 2048);
  cvt_to((const float*)d_in[I_TEXT],  x_txt, (size_t)BB * 768);
  cvt_to((const float*)d_in[I_META],  x_meta, (size_t)BB * 128);

  // ---- image path: 2048 -> 512 -> 2048 -> 512 -> 256 ----
  gemm(x_img, 2048, w_ae1, bias(I_AE1B), t1, 512, 0, 512, 2048, false, true);
  __bf16* t2 = x_img;  // x_img dead after g1
  gemm(t1, 512, w_ae2, bias(I_AE2B), t2, 2048, 0, 2048, 512, false, true);
  __bf16* t3 = t1;     // t1 dead after g2
  gemm(t2, 2048, w_ci1, bias(I_CI1B), t3, 512, 0, 512, 2048, false, true);
  gemm(t3, 512, w_ci2, bias(I_CI2B), imgv, 256, 0, 256, 512, false, true);
  // ---- text path ----
  gemm(x_txt, 768, w_fl, bias(I_FLB), txtv, 256, 0, 256, 768, false, true);
  // ---- projections ----
  gemm(txtv, 256, w_sl, bias(I_SLB), tsti, 256, 0, 128, 256, false, true);      // ts
  gemm(imgv, 256, w_sv, bias(I_SVB), ispii, 256, 0, 128, 256, false, true);     // isp
  gemm(txtv, 256, w_inv, bias(I_INVB), tsti, 256, 128, 128, 256, false, true);  // ti
  gemm(imgv, 256, w_inv, bias(I_INVB), ispii, 256, 128, 128, 256, false, true); // ii

  // ---- qkv_s (relu) into the recycled x_img region ----
  __bf16* tq = t2 + (size_t)0 * BB * 256;
  __bf16* tk = t2 + (size_t)1 * BB * 256;
  __bf16* tv = t2 + (size_t)2 * BB * 256;
  __bf16* iq = t2 + (size_t)3 * BB * 256;
  __bf16* ik = t2 + (size_t)4 * BB * 256;
  __bf16* iv = t2 + (size_t)5 * BB * 256;
  gemm(tsti, 256, w_qt, bias(I_QTB), tq, 256, 0, 256, 128, true, true);
  gemm(tsti, 256, w_kt, bias(I_KTB), tk, 256, 0, 256, 128, true, true);
  gemm(tsti, 256, w_vt, bias(I_VTB), tv, 256, 0, 256, 128, true, true);
  gemm(ispii, 256, w_qi, bias(I_QIB), iq, 256, 0, 256, 128, true, true);
  gemm(ispii, 256, w_ki, bias(I_KIB), ik, 256, 0, 256, 128, true, true);
  gemm(ispii, 256, w_vi, bias(I_VIB), iv, 256, 0, 256, 128, true, true);

  // ---- elementwise softmax attn (32-wide, 8 heads) -> static ----
  {
    long nG = (long)BB * 8;
    unsigned blk = (unsigned)((nG + 255) / 256);
    const float SCALE_S = 0.0625f;  // 256^-0.5
    softmax_mul32<__bf16><<<blk, 256, 0, stream>>>(tq, ik, iv, statb, 256, 768, 0, SCALE_S, 8, nG);
    softmax_mul32<__bf16><<<blk, 256, 0, stream>>>(iq, tk, tv, statb, 256, 768, 256, SCALE_S, 8, nG);
  }
  {
    long n = (long)BB * 128;
    unsigned blk = (unsigned)((n + 255) / 256);
    copy_cols<<<blk, 256, 0, stream>>>(tsti, 256, 128, statb, 768, 512, 128, BB);   // ti
    copy_cols<<<blk, 256, 0, stream>>>(ispii, 256, 128, statb, 768, 640, 128, BB);  // ii
  }

  // ---- qkv_m (no relu) ----
  __bf16* mq = t1 + (size_t)0 * BB * 128;  // t1/t3 region dead now
  __bf16* mk = t1 + (size_t)1 * BB * 128;
  __bf16* mv = t1 + (size_t)2 * BB * 128;
  gemm(x_meta, 128, w_qm, bias(I_QMB), mq, 128, 0, 128, 128, false, true);
  gemm(x_meta, 128, w_km, bias(I_KMB), mk, 128, 0, 128, 128, false, true);
  gemm(x_meta, 128, w_vm, bias(I_VMB), mv, 128, 0, 128, 128, false, true);
  __bf16* sq = x_txt + (size_t)0 * BB * 128;  // x_txt region dead now
  __bf16* sk = x_txt + (size_t)1 * BB * 128;
  __bf16* sv = x_txt + (size_t)2 * BB * 128;
  gemm(statb, 768, w_qs, bias(I_QSB), sq, 128, 0, 128, 768, false, true);
  gemm(statb, 768, w_ks, bias(I_KSB), sk, 128, 0, 128, 768, false, true);
  gemm(statb, 768, w_vs, bias(I_VSB), sv, 128, 0, 128, 768, false, true);

  // ---- meta/static cross attn (32-wide, 4 heads) -> outbuf (f32) ----
  {
    long nG = (long)BB * 4;
    unsigned blk = (unsigned)((nG + 255) / 256);
    const float SCALE_M = 0.08838834764831845f;  // 128^-0.5
    softmax_mul32<float><<<blk, 256, 0, stream>>>(mq, sk, sv, outbuf, 128, 256, 0, SCALE_M, 4, nG);
    softmax_mul32<float><<<blk, 256, 0, stream>>>(sq, mk, mv, outbuf, 128, 256, 128, SCALE_M, 4, nG);
  }

  // ---- LayerNorm(256) -> normed (bf16) ----
  layernorm_rows<<<(BB + 255) / 256, 256, 0, stream>>>(
      outbuf, (const float*)d_in[I_LNG], (const float*)d_in[I_LNB], normed, 256, BB);

  // ---- head: 256 -> 64 (relu, f32) -> 1 ----
  gemm(normed, 256, w_h1, bias(I_H1B), h1out, 64, 0, 64, 256, true, false);
  float* fout = (float*)d_out;
  head_logits<<<(BB + 255) / 256, 256, 0, stream>>>(
      h1out, (const float*)d_in[I_H2W], (const float*)d_in[I_H2B], fout, BB);

  // ---- recon heads (f32 straight into d_out) ----
  float* recon_t = fout + BB;
  float* recon_i = fout + BB + (size_t)BB * 256;
  gemm(tsti, 256, w_tr, bias(I_TRB), recon_t, 256, 0, 256, 256, false, false);
  gemm(ispii, 256, w_ir, bias(I_IRB), recon_i, 256, 0, 256, 256, false, false);
}
